// Encoder_65481071410993
// MI455X (gfx1250) — matrix operand order repaired and checked by reading, not compiled, not run on hardware
//
#include <hip/hip_runtime.h>
#include <math.h>

// ---------------------------------------------------------------------------
// HGT encoder forward for MI455X (gfx1250, wave32, WMMA).
// Dense GEMMs -> v_wmma_f32_16x16x32_f16; LDS tiles laid out so every WMMA
// fragment is fetched with ds_load_b128 (A row-major, B col-major in LDS).
// Edge message passing -> wave-per-edge, lane-per-channel (D==32==wave32),
// segment softmax via ordered-uint atomicMax + f32 global atomics.
// ---------------------------------------------------------------------------

typedef __attribute__((ext_vector_type(16))) _Float16 v16h;
typedef __attribute__((ext_vector_type(8)))  _Float16 v8h;
typedef __attribute__((ext_vector_type(4)))  _Float16 v4h;
typedef __attribute__((ext_vector_type(8)))  float    v8f;

static constexpr int   kNG    = 100000;
static constexpr int   kND    = 50000;
static constexpr int   kCIN   = 128;
static constexpr int   kH     = 8;
static constexpr int   kD     = 32;
static constexpr int   kHID   = 256;   // kH * kD
static constexpr int   kE     = 200000;
static constexpr float kSCALE = 0.17677669529663687f;  // 1/sqrt(32)

static constexpr int TM   = 64;   // block rows
static constexpr int TN   = 64;   // block cols (2 wave-cols x 32)
static constexpr int LDSS = 56;   // LDS row stride in halfs (112B: 16B-aligned,
                                  // 28*m mod 64 distinct banks for m=0..15)

// Monotone float -> uint key so unsigned atomicMax orders like float compare.
__device__ __forceinline__ unsigned fkey(float f) {
  int i = __float_as_int(f);
  return (unsigned)(i ^ ((i >> 31) | 0x80000000));
}
__device__ __forceinline__ float funkey(unsigned u) {
  int i = (u & 0x80000000u) ? (int)(u ^ 0x80000000u) : ~(int)u;
  return __int_as_float(i);
}
__device__ __forceinline__ bool finitef(float f) {
  return (__float_as_uint(f) & 0x7F800000u) != 0x7F800000u;
}

// ---------------------------------------------------------------------------
// C[M,N] = A[M,K] @ B[K,N] + bias[N]     (f32 in/out, f16 WMMA, f32 accum)
// Block: 256 threads = 8 waves as 4 (rows) x 2 (cols); each wave owns a
// 16x32 output strip = two 16x16 accumulators sharing one A fragment.
// ---------------------------------------------------------------------------
__global__ __launch_bounds__(256) void gemm_bias_kernel(
    const float* __restrict__ A, const float* __restrict__ B,
    const float* __restrict__ bias, float* __restrict__ C,
    int M, int K, int N) {
  __shared__ __align__(16) _Float16 As[TM][LDSS];   // [row][k]
  __shared__ __align__(16) _Float16 BsT[TN][LDSS];  // [col][k]  (transposed)

  const int tid  = threadIdx.x;
  const int lane = tid & 31;
  const int wave = tid >> 5;
  const int wr   = wave >> 1;               // 0..3
  const int wc   = wave & 1;                // 0..1
  const int rowBase = blockIdx.x * TM;
  const int colBase = blockIdx.y * TN;

  v8f acc0 = {};
  v8f acc1 = {};

  for (int kk = 0; kk < K; kk += 32) {
    // --- Stage A tile (TM x 32): float4 global loads -> packed v4h stores.
    for (int i = tid; i < TM * 32 / 4; i += 256) {          // 2 iters
      int r = i >> 3, c4 = (i & 7) << 2;
      int gr = rowBase + r;
      float4 v = make_float4(0.f, 0.f, 0.f, 0.f);
      if (gr < M) v = *(const float4*)&A[(size_t)gr * K + kk + c4];
      v4h h = {(_Float16)v.x, (_Float16)v.y, (_Float16)v.z, (_Float16)v.w};
      *(v4h*)&As[r][c4] = h;
    }
    // --- Stage B tile (32 x TN) transposed: float4 loads, b16 scatter.
    for (int i = tid; i < 32 * TN / 4; i += 256) {          // 2 iters
      int r = i >> 4, c4 = (i & 15) << 2;
      float4 v = *(const float4*)&B[(size_t)(kk + r) * N + colBase + c4];
      BsT[c4 + 0][r] = (_Float16)v.x;
      BsT[c4 + 1][r] = (_Float16)v.y;
      BsT[c4 + 2][r] = (_Float16)v.z;
      BsT[c4 + 3][r] = (_Float16)v.w;
    }
    __syncthreads();

    // A fragment: lane m holds row M=m%16,
    // K = akb..akb+15 contiguous (matches B)  -> two ds_load_b128.
    const int am  = lane & 15;
    const int akb = (lane < 16) ? 0 : 16;
    v8h a0 = *(const v8h*)&As[wr * 16 + am][akb];
    v8h a1 = *(const v8h*)&As[wr * 16 + am][akb + 8];
    v16h af = __builtin_shufflevector(a0, a1, 0, 1, 2, 3, 4, 5, 6, 7,
                                      8, 9, 10, 11, 12, 13, 14, 15);

    // B fragments: lane holds col N=lane%16, K = bkb..bkb+15 contiguous.
    const int bkb = (lane < 16) ? 0 : 16;
    const int bc0 = wc * 32 + (lane & 15);
    const int bc1 = bc0 + 16;
    const v8h* bp0 = (const v8h*)&BsT[bc0][bkb];
    const v8h* bp1 = (const v8h*)&BsT[bc1][bkb];
    v16h bf0 = __builtin_shufflevector(bp0[0], bp0[1], 0, 1, 2, 3, 4, 5, 6, 7,
                                       8, 9, 10, 11, 12, 13, 14, 15);
    v16h bf1 = __builtin_shufflevector(bp1[0], bp1[1], 0, 1, 2, 3, 4, 5, 6, 7,
                                       8, 9, 10, 11, 12, 13, 14, 15);

    acc0 = __builtin_amdgcn_wmma_f32_16x16x32_f16(false, af, false, bf0,
                                                  (short)0, acc0, false, false);
    acc1 = __builtin_amdgcn_wmma_f32_16x16x32_f16(false, af, false, bf1,
                                                  (short)0, acc1, false, false);
    __syncthreads();
  }

  // C/D layout: lanes 0-15: N=lane, M=j; lanes 16-31: N=lane-16, M=j+8.
  const int n0    = colBase + wc * 32 + (lane & 15);
  const int mrow0 = rowBase + wr * 16 + ((lane < 16) ? 0 : 8);
  const float bv0 = bias[n0];
  const float bv1 = bias[n0 + 16];
#pragma unroll
  for (int j = 0; j < 8; ++j) {
    int row = mrow0 + j;
    if (row < M) {
      C[(size_t)row * N + n0]      = acc0[j] + bv0;
      C[(size_t)row * N + n0 + 16] = acc1[j] + bv1;
    }
  }
}

// ---------------------------------------------------------------------------
// Edge pass 1: logits[e,h] = (q[dst] . (k[src] @ a_rel[h])) * p[h] * SCALE
// plus segment max via ordered-uint atomicMax. One wave per edge, lane = f.
// ---------------------------------------------------------------------------
__global__ __launch_bounds__(256) void edge_logits_kernel(
    const float* __restrict__ Ksrc, const float* __restrict__ Qdst,
    const int* __restrict__ src, const int* __restrict__ dst,
    const float* __restrict__ a_rel, const float* __restrict__ p_rel,
    float* __restrict__ logits, unsigned* __restrict__ mkey, int nEdges) {
  const int lane = threadIdx.x & 31;
  const int e    = blockIdx.x * 8 + (threadIdx.x >> 5);
  if (e >= nEdges) return;
  const int s = src[e], d = dst[e];
  const float* krow = Ksrc + (size_t)s * kHID;
  const float* qrow = Qdst + (size_t)d * kHID;
  for (int h = 0; h < kH; ++h) {
    const float kv = krow[h * kD + lane];           // lane holds k[.,h,lane]
    const float* Am = a_rel + h * kD * kD;
    float ke = 0.0f;                                // k_e[f = lane]
#pragma unroll
    for (int dd = 0; dd < kD; ++dd) {
      float kd_ = __shfl(kv, dd, 32);
      ke += kd_ * Am[dd * kD + lane];
    }
    float t = ke * qrow[h * kD + lane];
#pragma unroll
    for (int off = 16; off > 0; off >>= 1) t += __shfl_xor(t, off, 32);
    float logit = t * p_rel[h] * kSCALE;
    if (lane == 0) {
      logits[(size_t)e * kH + h] = logit;
      atomicMax(&mkey[(size_t)d * kH + h], fkey(logit));
    }
  }
}

// Edge pass 2: e = exp(logit - m[dst]); overwrite logits with e; atomic sum.
__global__ __launch_bounds__(256) void edge_expsum_kernel(
    float* __restrict__ logits, const int* __restrict__ dst,
    const unsigned* __restrict__ mkey, float* __restrict__ ssum, int total) {
  int idx = blockIdx.x * 256 + threadIdx.x;
  if (idx >= total) return;
  int e = idx / kH, h = idx - e * kH;
  int d = dst[e];
  float m = funkey(mkey[(size_t)d * kH + h]);
  if (!finitef(m)) m = 0.0f;                        // empty-segment guard
  float ex = expf(logits[idx] - m);
  logits[idx] = ex;
  __hip_atomic_fetch_add(&ssum[(size_t)d * kH + h], ex,
                         __ATOMIC_RELAXED, __HIP_MEMORY_SCOPE_AGENT);
}

// Edge pass 3: v_e = v[src] @ m_rel[h]; out[dst] += alpha * v_e (f32 atomics).
__global__ __launch_bounds__(256) void edge_scatter_kernel(
    const float* __restrict__ Vsrc, const int* __restrict__ src,
    const int* __restrict__ dst, const float* __restrict__ m_rel,
    const float* __restrict__ logits, const float* __restrict__ ssum,
    float* __restrict__ outp, int nEdges) {
  const int lane = threadIdx.x & 31;
  const int e    = blockIdx.x * 8 + (threadIdx.x >> 5);
  if (e >= nEdges) return;
  const int s = src[e], d = dst[e];
  const float* vrow = Vsrc + (size_t)s * kHID;
  for (int h = 0; h < kH; ++h) {
    const float vv = vrow[h * kD + lane];
    const float* Mm = m_rel + h * kD * kD;
    float ve = 0.0f;
#pragma unroll
    for (int dd = 0; dd < kD; ++dd) {
      float vd_ = __shfl(vv, dd, 32);
      ve += vd_ * Mm[dd * kD + lane];
    }
    float alpha = logits[(size_t)e * kH + h] /
                  (ssum[(size_t)d * kH + h] + 1e-16f);
    __hip_atomic_fetch_add(&outp[(size_t)d * kHID + h * kD + lane], alpha * ve,
                           __ATOMIC_RELAXED, __HIP_MEMORY_SCOPE_AGENT);
  }
}

// ---------------------------------------------------------------------------
// Elementwise helpers
// ---------------------------------------------------------------------------
__global__ __launch_bounds__(256) void fill_zero_kernel(unsigned* __restrict__ p,
                                                        size_t n) {
  size_t i = (size_t)blockIdx.x * 256 + threadIdx.x;
  if (i < n) p[i] = 0u;
}

__global__ __launch_bounds__(256) void gelu_kernel(const float* __restrict__ in,
                                                   float* __restrict__ out,
                                                   size_t n) {
  size_t i = (size_t)blockIdx.x * 256 + threadIdx.x;
  if (i < n) {
    float x = in[i];
    out[i] = 0.5f * x * (1.0f + erff(x * 0.70710678118654752f));
  }
}

// h = beta*h + (1-beta)*x_proj; PReLU per channel.
__global__ __launch_bounds__(256) void finish_kernel(
    const float* __restrict__ hbuf, const float* __restrict__ xproj,
    const float* __restrict__ skip, const float* __restrict__ prelu,
    float* __restrict__ out, size_t n) {
  size_t i = (size_t)blockIdx.x * 256 + threadIdx.x;
  if (i < n) {
    float beta = 1.0f / (1.0f + expf(-skip[0]));
    float h = beta * hbuf[i] + (1.0f - beta) * xproj[i];
    float p = prelu[i & (kHID - 1)];
    out[i] = (h > 0.0f) ? h : p * h;
  }
}

// ---------------------------------------------------------------------------
// Host launcher
// ---------------------------------------------------------------------------
extern "C" void kernel_launch(void* const* d_in, const int* in_sizes, int n_in,
                              void* d_out, int out_size, void* d_ws, size_t ws_size,
                              hipStream_t stream) {
  (void)in_sizes; (void)n_in; (void)out_size; (void)ws_size;

  const float* x_gene    = (const float*)d_in[0];
  const float* x_disease = (const float*)d_in[1];
  const float* Win_g = (const float*)d_in[2];  const float* bin_g = (const float*)d_in[3];
  const float* Wk_g  = (const float*)d_in[4];  const float* bk_g  = (const float*)d_in[5];
  const float* Wq_g  = (const float*)d_in[6];  const float* bq_g  = (const float*)d_in[7];
  const float* Wv_g  = (const float*)d_in[8];  const float* bv_g  = (const float*)d_in[9];
  const float* Wa_g  = (const float*)d_in[10]; const float* ba_g  = (const float*)d_in[11];
  const float* skip_g = (const float*)d_in[12];
  const float* Win_d = (const float*)d_in[13]; const float* bin_d = (const float*)d_in[14];
  const float* Wk_d  = (const float*)d_in[15]; const float* bk_d  = (const float*)d_in[16];
  const float* Wq_d  = (const float*)d_in[17]; const float* bq_d  = (const float*)d_in[18];
  const float* Wv_d  = (const float*)d_in[19]; const float* bv_d  = (const float*)d_in[20];
  const float* Wa_d  = (const float*)d_in[21]; const float* ba_d  = (const float*)d_in[22];
  const float* skip_d = (const float*)d_in[23];
  const float* a_g2d = (const float*)d_in[24]; const float* m_g2d = (const float*)d_in[25];
  const float* p_g2d = (const float*)d_in[26];
  const float* a_d2g = (const float*)d_in[27]; const float* m_d2g = (const float*)d_in[28];
  const float* p_d2g = (const float*)d_in[29];
  const float* a_g2g = (const float*)d_in[30]; const float* m_g2g = (const float*)d_in[31];
  const float* p_g2g = (const float*)d_in[32];
  const float* prelu_w = (const float*)d_in[33];
  const int* g2d_src = (const int*)d_in[34];   const int* g2d_dst = (const int*)d_in[35];
  const int* d2g_src = (const int*)d_in[36];   const int* d2g_dst = (const int*)d_in[37];
  const int* g2g_src = (const int*)d_in[38];   const int* g2g_dst = (const int*)d_in[39];

  // Workspace layout (floats). Total ~195.2M floats (~781 MB).
  const size_t NGH = (size_t)kNG * kHID;   // 25.6M
  const size_t NDH = (size_t)kND * kHID;   // 12.8M
  const size_t maxNH = (size_t)kNG * kH;   // 0.8M  (max over dst types)
  float* ws   = (float*)d_ws;
  float* xg   = ws;
  float* xd   = xg  + NGH;
  float* kg   = xd  + NDH;   // later reused as gelu(out_g)
  float* qg   = kg  + NGH;   // later reused as final-GEMM output (gene)
  float* vg   = qg  + NGH;
  float* kd   = vg  + NGH;   // later reused as gelu(out_d)
  float* qd   = kd  + NDH;   // later reused as final-GEMM output (disease)
  float* vd   = qd  + NDH;
  float* outg = vd  + NDH;
  float* outd = outg + NGH;  // contiguous with outg
  float* logits = outd + NDH;                       // kE * kH
  unsigned* mkey = (unsigned*)(logits + (size_t)kE * kH);  // maxNH
  float* ssum = (float*)(mkey + maxNH);             // maxNH (contiguous w/ mkey)

  float* out_gene = (float*)d_out;
  float* out_dis  = out_gene + NGH;

  auto blocks1d = [](size_t n) { return (unsigned)((n + 255) / 256); };
  auto gemmGrid = [](int M, int N) { return dim3((M + TM - 1) / TM, N / TN); };

  // Zero message accumulators (outg|outd contiguous).
  fill_zero_kernel<<<blocks1d(NGH + NDH), 256, 0, stream>>>((unsigned*)outg,
                                                            NGH + NDH);

  // Input projections.
  gemm_bias_kernel<<<gemmGrid(kNG, kHID), 256, 0, stream>>>(
      x_gene, Win_g, bin_g, xg, kNG, kCIN, kHID);
  gemm_bias_kernel<<<gemmGrid(kND, kHID), 256, 0, stream>>>(
      x_disease, Win_d, bin_d, xd, kND, kCIN, kHID);

  // K/Q/V projections.
  gemm_bias_kernel<<<gemmGrid(kNG, kHID), 256, 0, stream>>>(xg, Wk_g, bk_g, kg, kNG, kHID, kHID);
  gemm_bias_kernel<<<gemmGrid(kNG, kHID), 256, 0, stream>>>(xg, Wq_g, bq_g, qg, kNG, kHID, kHID);
  gemm_bias_kernel<<<gemmGrid(kNG, kHID), 256, 0, stream>>>(xg, Wv_g, bv_g, vg, kNG, kHID, kHID);
  gemm_bias_kernel<<<gemmGrid(kND, kHID), 256, 0, stream>>>(xd, Wk_d, bk_d, kd, kND, kHID, kHID);
  gemm_bias_kernel<<<gemmGrid(kND, kHID), 256, 0, stream>>>(xd, Wq_d, bq_d, qd, kND, kHID, kHID);
  gemm_bias_kernel<<<gemmGrid(kND, kHID), 256, 0, stream>>>(xd, Wv_d, bv_d, vd, kND, kHID, kHID);

  // Message passing per relation (sequential; mkey/ssum/logits reused).
  const unsigned edgeBlocks = (kE + 7) / 8;
  auto run_rel = [&](const float* Ksrc, const float* Vsrc, const float* Qdst,
                     const int* src, const int* dst, const float* a_r,
                     const float* m_r, const float* p_r, float* outp) {
    fill_zero_kernel<<<blocks1d(2 * maxNH), 256, 0, stream>>>(mkey, 2 * maxNH);
    edge_logits_kernel<<<edgeBlocks, 256, 0, stream>>>(
        Ksrc, Qdst, src, dst, a_r, p_r, logits, mkey, kE);
    edge_expsum_kernel<<<blocks1d((size_t)kE * kH), 256, 0, stream>>>(
        logits, dst, mkey, ssum, kE * kH);
    edge_scatter_kernel<<<edgeBlocks, 256, 0, stream>>>(
        Vsrc, src, dst, m_r, logits, ssum, outp, kE);
  };
  run_rel(kg, vg, qd, g2d_src, g2d_dst, a_g2d, m_g2d, p_g2d, outd);
  run_rel(kd, vd, qg, d2g_src, d2g_dst, a_d2g, m_d2g, p_d2g, outg);
  run_rel(kg, vg, qg, g2g_src, g2g_dst, a_g2g, m_g2g, p_g2g, outg);

  // Output head: gelu -> GEMM -> skip blend + PReLU.
  gelu_kernel<<<blocks1d(NGH), 256, 0, stream>>>(outg, kg, NGH);  // reuse kg
  gelu_kernel<<<blocks1d(NDH), 256, 0, stream>>>(outd, kd, NDH);  // reuse kd
  gemm_bias_kernel<<<gemmGrid(kNG, kHID), 256, 0, stream>>>(kg, Wa_g, ba_g, qg, kNG, kHID, kHID);
  gemm_bias_kernel<<<gemmGrid(kND, kHID), 256, 0, stream>>>(kd, Wa_d, ba_d, qd, kND, kHID, kHID);
  finish_kernel<<<blocks1d(NGH), 256, 0, stream>>>(qg, xg, skip_g, prelu_w, out_gene, NGH);
  finish_kernel<<<blocks1d(NDH), 256, 0, stream>>>(qd, xd, skip_d, prelu_w, out_dis, NDH);
}
